// SwinSelfAttention_7980049236528
// MI455X (gfx1250) — matrix operand order repaired and checked by reading, not compiled, not run on hardware
//
#include <hip/hip_runtime.h>

typedef __attribute__((ext_vector_type(16))) _Float16 v16h;
typedef __attribute__((ext_vector_type(8)))  _Float16 v8h;
typedef __attribute__((ext_vector_type(4)))  _Float16 v4h;
typedef __attribute__((ext_vector_type(8)))  float    v8f;

// ---- LDS layout (bytes). Total 311296 = 304 KB <= 320 KB/WGP ----
#define OFF_X    0        // X stage [64][520] f16 ; later probs [8][64][72] f16
#define OFF_Q    73728    // Q [8][64][72] f16     ; later rep [64][520] f16
#define OFF_K    147456   // K [8][64][72] f16
#define OFF_V    221184   // V^T [8][64][72] f16
#define OFF_ADD  294912   // addend (bias - inf*mask) [64][64] f32
#define LDS_TOTAL 311296

#define XSTRIDE 520   // halves; 1040 B rows -> 16B aligned fragments
#define HSTRIDE 72    // halves; 144 B rows  -> 16B aligned, conflict-free

__device__ __forceinline__ v8f wmma_f16(v16h a, v16h b, v8f c) {
  return __builtin_amdgcn_wmma_f32_16x16x32_f16(false, a, false, b, (short)0, c,
                                                false, false);
}

// A fragment (M=16,K=32) from row-major f16 LDS [m][k], stride S halves.
// lanes 0-15: M rows, K=k0..k0+15 contiguous
// lanes 16-31: same rows, K=k0+16..k0+31.
__device__ __forceinline__ v16h load_a_lds(const _Float16* buf, int S, int m0,
                                           int k0, int lane) {
  int m  = m0 + (lane & 15);
  int kb = k0 + 16 * (lane >> 4);
  const _Float16* p = buf + m * S + kb;
  v8h lo = *(const v8h*)(p);
  v8h hi = *(const v8h*)(p + 8);
  return __builtin_shufflevector(lo, hi, 0, 1, 2, 3, 4, 5, 6, 7, 8, 9, 10, 11,
                                 12, 13, 14, 15);
}

// B fragment (K=32,N=16) where LDS buffer stores B transposed: buf[n][k].
// lane = n (0-15) with K=k0..k0+15 ; lanes 16-31: K=k0+16..k0+31.
__device__ __forceinline__ v16h load_bT_lds(const _Float16* buf, int S, int n0,
                                            int k0, int lane) {
  int n  = n0 + (lane & 15);
  int kb = k0 + 16 * (lane >> 4);
  const _Float16* p = buf + n * S + kb;
  v8h lo = *(const v8h*)(p);
  v8h hi = *(const v8h*)(p + 8);
  return __builtin_shufflevector(lo, hi, 0, 1, 2, 3, 4, 5, 6, 7, 8, 9, 10, 11,
                                 12, 13, 14, 15);
}

// B fragment from global f32 weight stored row-major W[n][k] (ldw floats),
// converting to f16 on the fly. 16 consecutive floats per lane (4x b128).
__device__ __forceinline__ v16h load_bT_gf32(const float* W, int ldw, int n0,
                                             int k0, int lane) {
  int n  = n0 + (lane & 15);
  int kb = k0 + 16 * (lane >> 4);
  const float* p = W + n * ldw + kb;
  v16h b;
#pragma unroll
  for (int i = 0; i < 16; i += 4) {
    float4 f = *(const float4*)(p + i);
    b[i + 0] = (_Float16)f.x;
    b[i + 1] = (_Float16)f.y;
    b[i + 2] = (_Float16)f.z;
    b[i + 3] = (_Float16)f.w;
  }
  return b;
}

__global__ __launch_bounds__(256)
void swin_attn_kernel(const float* __restrict__ patches,
                      const float* __restrict__ Wq, const float* __restrict__ bq,
                      const float* __restrict__ Wk, const float* __restrict__ bk,
                      const float* __restrict__ Wv, const float* __restrict__ bv,
                      const float* __restrict__ Wo, const float* __restrict__ bo,
                      const float* __restrict__ pos_bias,
                      const unsigned char* __restrict__ mask,
                      const int* __restrict__ idx_y,
                      const int* __restrict__ idx_x,
                      float* __restrict__ out) {
  extern __shared__ char smem[];
  _Float16* Xs  = (_Float16*)(smem + OFF_X);
  _Float16* Qs  = (_Float16*)(smem + OFF_Q);
  _Float16* Ks  = (_Float16*)(smem + OFF_K);
  _Float16* Vt  = (_Float16*)(smem + OFF_V);
  float*    Add = (float*)(smem + OFF_ADD);

  const int tid  = threadIdx.x;
  const int lane = tid & 31;
  const int lo   = lane & 15;
  const int hi   = lane >> 4;
  const int w    = tid >> 5;            // wave id == head id
  const int win  = blockIdx.x;          // window index: b*256 + y*16 + x
  const int yx   = win & 255;           // y*16 + x (mask index)
  const size_t winbase = (size_t)win * (64 * 512);

  // ---------------- Phase 1: stage X (f32->f16) and bias/mask addend -------
  {
    const float4* Xg4 = (const float4*)(patches + winbase);
    for (int i = tid; i < (64 * 512) / 4; i += 256) {
      int r = i >> 7;             // 128 float4 per row
      int c = (i & 127) << 2;
      float4 f = Xg4[i];
      v4h h4 = {(_Float16)f.x, (_Float16)f.y, (_Float16)f.z, (_Float16)f.w};
      *(v4h*)(Xs + r * XSTRIDE + c) = h4;
    }
    const unsigned char* mk = mask + (size_t)yx * 4096;
    const float NEG_INF = -__builtin_inff();
    for (int i = tid; i < 4096; i += 256) {
      float bval = pos_bias[idx_y[i] * 15 + idx_x[i]];
      Add[i] = mk[i] ? NEG_INF : bval;
    }
  }
  __syncthreads();

  // ---------------- Phase 2: Q,K,V projections for head w ------------------
  _Float16* Qh = Qs + w * (64 * HSTRIDE);
  _Float16* Kh = Ks + w * (64 * HSTRIDE);
  _Float16* Vh = Vt + w * (64 * HSTRIDE);
  {
    const float* Wqh = Wq + (size_t)w * (64 * 512);
    const float* Wkh = Wk + (size_t)w * (64 * 512);
    const float* Wvh = Wv + (size_t)w * (64 * 512);
    const v8f z = {};
#pragma unroll
    for (int nt = 0; nt < 4; ++nt) {
      v8f aq[4], ak[4], av[4];
#pragma unroll
      for (int mt = 0; mt < 4; ++mt) { aq[mt] = z; ak[mt] = z; av[mt] = z; }
      for (int kk = 0; kk < 16; ++kk) {
        int k0 = kk * 32;
        v16h fq = load_bT_gf32(Wqh, 512, nt * 16, k0, lane);
        v16h fk = load_bT_gf32(Wkh, 512, nt * 16, k0, lane);
        v16h fv = load_bT_gf32(Wvh, 512, nt * 16, k0, lane);
#pragma unroll
        for (int mt = 0; mt < 4; ++mt) {
          v16h a = load_a_lds(Xs, XSTRIDE, mt * 16, k0, lane);
          aq[mt] = wmma_f16(a, fq, aq[mt]);
          ak[mt] = wmma_f16(a, fk, ak[mt]);
          av[mt] = wmma_f16(a, fv, av[mt]);
        }
      }
      float vbq = bq[w * 64 + nt * 16 + lo];
      float vbk = bk[w * 64 + nt * 16 + lo];
      float vbv = bv[w * 64 + nt * 16 + lo];
      int col = nt * 16 + lo;
#pragma unroll
      for (int mt = 0; mt < 4; ++mt) {
#pragma unroll
        for (int r = 0; r < 8; ++r) {
          int row = mt * 16 + r + 8 * hi;
          // fold 1/sqrt(D)=0.125 into Q (exact power of two)
          Qh[row * HSTRIDE + col] = (_Float16)((aq[mt][r] + vbq) * 0.125f);
          Kh[row * HSTRIDE + col] = (_Float16)(ak[mt][r] + vbk);
          Vh[col * HSTRIDE + row] = (_Float16)(av[mt][r] + vbv);  // transposed
        }
      }
    }
  }
  __syncthreads();  // all waves done reading Xs; probs region reuses it

  // ---------------- Phase 3a: scores + softmax -> probs (f16) --------------
  _Float16* Ph = Xs + w * (64 * HSTRIDE);
  {
    const v8f z = {};
#pragma unroll
    for (int mt = 0; mt < 4; ++mt) {
      v8f sc[4];
#pragma unroll
      for (int nt = 0; nt < 4; ++nt) {
        sc[nt] = z;
#pragma unroll
        for (int kk = 0; kk < 2; ++kk) {
          v16h a = load_a_lds(Qh, HSTRIDE, mt * 16, kk * 32, lane);
          v16h b = load_bT_lds(Kh, HSTRIDE, nt * 16, kk * 32, lane);
          sc[nt] = wmma_f16(a, b, sc[nt]);
        }
#pragma unroll
        for (int r = 0; r < 8; ++r)
          sc[nt][r] += Add[(mt * 16 + r + 8 * hi) * 64 + nt * 16 + lo];
      }
#pragma unroll
      for (int r = 0; r < 8; ++r) {
        float m = fmaxf(fmaxf(sc[0][r], sc[1][r]), fmaxf(sc[2][r], sc[3][r]));
        m = fmaxf(m, __shfl_xor(m, 1, 16));
        m = fmaxf(m, __shfl_xor(m, 2, 16));
        m = fmaxf(m, __shfl_xor(m, 4, 16));
        m = fmaxf(m, __shfl_xor(m, 8, 16));
        float e0 = __expf(sc[0][r] - m);
        float e1 = __expf(sc[1][r] - m);
        float e2 = __expf(sc[2][r] - m);
        float e3 = __expf(sc[3][r] - m);
        float s = e0 + e1 + e2 + e3;
        s += __shfl_xor(s, 1, 16);
        s += __shfl_xor(s, 2, 16);
        s += __shfl_xor(s, 4, 16);
        s += __shfl_xor(s, 8, 16);
        float inv = 1.0f / s;
        int row = (mt * 16 + r + 8 * hi) * HSTRIDE;
        Ph[row + 0 * 16 + lo] = (_Float16)(e0 * inv);
        Ph[row + 1 * 16 + lo] = (_Float16)(e1 * inv);
        Ph[row + 2 * 16 + lo] = (_Float16)(e2 * inv);
        Ph[row + 3 * 16 + lo] = (_Float16)(e3 * inv);
      }
    }
  }
  __syncthreads();  // all waves done reading Q region; rep reuses it

  // ---------------- Phase 3b: rep = probs @ V  (writes rep[64][520]) -------
  {
    _Float16* rep = Qs;  // stride XSTRIDE; head w owns cols [64w, 64w+64)
    const v8f z = {};
#pragma unroll
    for (int nt = 0; nt < 4; ++nt) {
      v8f acc[4];
#pragma unroll
      for (int mt = 0; mt < 4; ++mt) acc[mt] = z;
#pragma unroll
      for (int kk = 0; kk < 2; ++kk) {
        v16h b = load_bT_lds(Vh, HSTRIDE, nt * 16, kk * 32, lane);
#pragma unroll
        for (int mt = 0; mt < 4; ++mt) {
          v16h a = load_a_lds(Ph, HSTRIDE, mt * 16, kk * 32, lane);
          acc[mt] = wmma_f16(a, b, acc[mt]);
        }
      }
      int col = w * 64 + nt * 16 + lo;
#pragma unroll
      for (int mt = 0; mt < 4; ++mt)
#pragma unroll
        for (int r = 0; r < 8; ++r)
          rep[(mt * 16 + r + 8 * hi) * XSTRIDE + col] = (_Float16)acc[mt][r];
    }
  }
  __syncthreads();  // every wave needs all heads' rep columns

  // ---------------- Phase 4: out = rep @ Wo^T + bo --------------------------
  {
    const _Float16* rep = Qs;
    float* og = out + winbase;
    const int n0 = w * 64;
    const v8f z = {};
#pragma unroll
    for (int nt = 0; nt < 4; ++nt) {
      v8f acc[4];
#pragma unroll
      for (int mt = 0; mt < 4; ++mt) acc[mt] = z;
      for (int kk = 0; kk < 16; ++kk) {
        int k0 = kk * 32;
        v16h b = load_bT_gf32(Wo, 512, n0 + nt * 16, k0, lane);
#pragma unroll
        for (int mt = 0; mt < 4; ++mt) {
          v16h a = load_a_lds(rep, XSTRIDE, mt * 16, k0, lane);
          acc[mt] = wmma_f16(a, b, acc[mt]);
        }
      }
      int col = n0 + nt * 16 + lo;
      float fb = bo[col];
#pragma unroll
      for (int mt = 0; mt < 4; ++mt)
#pragma unroll
        for (int r = 0; r < 8; ++r)
          og[(size_t)(mt * 16 + r + 8 * hi) * 512 + col] = acc[mt][r] + fb;
    }
  }
}

extern "C" void kernel_launch(void* const* d_in, const int* in_sizes, int n_in,
                              void* d_out, int out_size, void* d_ws,
                              size_t ws_size, hipStream_t stream) {
  (void)in_sizes; (void)n_in; (void)d_ws; (void)ws_size; (void)out_size;
  const float* patches  = (const float*)d_in[0];
  const float* Wq       = (const float*)d_in[1];
  const float* bq       = (const float*)d_in[2];
  const float* Wk       = (const float*)d_in[3];
  const float* bk       = (const float*)d_in[4];
  const float* Wv       = (const float*)d_in[5];
  const float* bv       = (const float*)d_in[6];
  const float* Wo       = (const float*)d_in[7];
  const float* bo       = (const float*)d_in[8];
  const float* pos_bias = (const float*)d_in[9];
  const unsigned char* mask = (const unsigned char*)d_in[10];
  const int* idx_y      = (const int*)d_in[11];
  const int* idx_x      = (const int*)d_in[12];
  float* outp           = (float*)d_out;

  static bool attr_set = false;
  // Allow > default dynamic LDS (304 KB of the 320 KB WGP budget).
  // Idempotent and capture-safe (host-side attribute, not a stream op).
  hipFuncSetAttribute((const void*)swin_attn_kernel,
                      hipFuncAttributeMaxDynamicSharedMemorySize, LDS_TOTAL);
  (void)attr_set;

  // 2048 windows (B=8 * 16 * 16); 256 threads = 8 wave32 (one head each).
  swin_attn_kernel<<<2048, 256, LDS_TOTAL, stream>>>(
      patches, Wq, bq, Wk, bk, Wv, bv, Wo, bo, pos_bias, mask, idx_y, idx_x,
      outp);
}
